// model_51565377356486
// MI455X (gfx1250) — matrix operand order repaired and checked by reading, not compiled, not run on hardware
//
#include <hip/hip_runtime.h>
#include <hip/hip_bf16.h>
#include <math.h>

// ---------------- model dims ----------------
#define WB   16      // batch
#define WS   256     // encoder seq len
#define WT   64      // decoder steps
#define WH   1024    // hidden
#define WEH  512     // encoder dir hidden
#define WV   32000   // vocab
#define NEGV (-1e30f)

typedef __attribute__((ext_vector_type(16))) __bf16 v16bf;
typedef __attribute__((ext_vector_type(8)))  __bf16 v8bf;
typedef __attribute__((ext_vector_type(4)))  __bf16 v4bf;
typedef __attribute__((ext_vector_type(8)))  float  v8f;
typedef __attribute__((ext_vector_type(4)))  float  v4f;

__device__ __forceinline__ float sig_(float x) { return 1.0f / (1.0f + expf(-x)); }

// ================= bf16 fragment loaders (pure bf16 path, no cvt) =================
// A-matrix 16x32 (MxK) bf16 layout (ISA 7.12.2):
//  lane L: row M=L&15; halves 0..7 -> K=c0..c0+7, halves 8..15 -> K=c0+16..c0+23, c0=8*(L/16)
__device__ __forceinline__ v16bf loadA_bf16(const __bf16* __restrict__ p, int ld, int lane) {
  const __bf16* q = p + (long)(lane & 15) * ld + ((lane >> 4) << 3);
  v8bf a0 = *(const v8bf*)(q);
  v8bf a1 = *(const v8bf*)(q + 16);
  return __builtin_shufflevector(a0, a1, 0,1,2,3,4,5,6,7,8,9,10,11,12,13,14,15);
}
// B-matrix 32x16 (KxN): B[k][n] = W[n][k], W row-major (N,K).
// lane L: n=L&15; halves 0..7 -> K=c0..c0+7, halves 8..15 -> K=c0+16..c0+23, c0=8*(L/16) (same as A).
__device__ __forceinline__ v16bf loadB_bf16(const __bf16* __restrict__ p, int ld, int lane) {
  const __bf16* q = p + (long)(lane & 15) * ld + ((lane >> 4) << 3);
  v8bf b0 = *(const v8bf*)(q);
  v8bf b1 = *(const v8bf*)(q + 16);
  return __builtin_shufflevector(b0, b1, 0,1,2,3,4,5,6,7,8,9,10,11,12,13,14,15);
}

// One wave: 16(M) x 64(N) slab, A fragment shared by 4 N-tiles. K % 32 == 0.
// C[m][n] = sum_k A[m][k]*W[n][k] (+Cin)(+bias) (tanh if ep==1) ; optional bf16 copy-out.
template <int TN>
__device__ __forceinline__ void gemm_wave_bf(const __bf16* __restrict__ A, const __bf16* __restrict__ W,
                                             const float* __restrict__ Cin, const float* __restrict__ bias,
                                             float* __restrict__ C, __bf16* __restrict__ Cbf,
                                             int K, int lda, int ldw, int ldcin, int ldc,
                                             int ty, int tx0, int lane, int ep) {
  v8f acc[TN];
#pragma unroll
  for (int j = 0; j < TN; ++j)
#pragma unroll
    for (int r = 0; r < 8; ++r) acc[j][r] = 0.0f;

  const __bf16* Arow = A + (long)ty * 16 * lda;
  for (int k = 0; k + 32 <= K; k += 32) {
    v16bf af = loadA_bf16(Arow + k, lda, lane);
#pragma unroll
    for (int j = 0; j < TN; ++j) {
      v16bf bf = loadB_bf16(W + (long)(tx0 + j) * 16 * ldw + k, ldw, lane);
      acc[j] = __builtin_amdgcn_wmma_f32_16x16x32_bf16(false, af, false, bf,
                                                       (short)0, acc[j], false, false);
    }
  }
  int n0 = lane & 15;
  int mh = (lane >> 4) << 3;
#pragma unroll
  for (int j = 0; j < TN; ++j) {
    int ncol = (tx0 + j) * 16 + n0;
#pragma unroll
    for (int r = 0; r < 8; ++r) {
      int m = ty * 16 + mh + r;
      float v = acc[j][r];
      if (Cin)  v += Cin[(long)m * ldcin + ncol];
      if (bias) v += bias[ncol];
      if (ep == 1) v = tanhf(v);
      C[(long)m * ldc + ncol] = v;
      if (Cbf) Cbf[(long)m * ldc + ncol] = (__bf16)v;
    }
  }
}

__global__ void gemm_bf16_xwT(const __bf16* __restrict__ A, const __bf16* __restrict__ W,
                              const float* __restrict__ Cin, const float* __restrict__ bias,
                              float* __restrict__ C, __bf16* __restrict__ Cbf, int M, int N, int K,
                              int lda, int ldw, int ldcin, int ldc, int ep) {
  int lane = threadIdx.x & 31;
  int wave = blockIdx.x * (blockDim.x >> 5) + (threadIdx.x >> 5);
  int ntx = N >> 6;
  int tx0 = (wave % ntx) * 4;
  int ty  = wave / ntx;
  if (ty * 16 >= M) return;  // wave-uniform
  gemm_wave_bf<4>(A, W, Cin, bias, C, Cbf, K, lda, ldw, ldcin, ldc, ty, tx0, lane, ep);
}

// ================= f32-input GEMM (on-the-fly cvt) — one-shot xproj GEMMs =================
struct RawFrag { v4f r[4]; };
__device__ __forceinline__ RawFrag loadA_raw(const float* __restrict__ p, int ld, int lane) {
  const float* q = p + (long)(lane & 15) * ld + ((lane >> 4) << 3);
  RawFrag x;
  x.r[0] = *(const v4f*)(q);      x.r[1] = *(const v4f*)(q + 4);
  x.r[2] = *(const v4f*)(q + 16); x.r[3] = *(const v4f*)(q + 20);
  return x;
}
__device__ __forceinline__ RawFrag loadB_raw(const float* __restrict__ p, int ld, int lane) {
  const float* q = p + (long)(lane & 15) * ld + ((lane >> 4) << 3);
  RawFrag x;
  x.r[0] = *(const v4f*)(q);      x.r[1] = *(const v4f*)(q + 4);
  x.r[2] = *(const v4f*)(q + 16); x.r[3] = *(const v4f*)(q + 20);
  return x;
}
__device__ __forceinline__ v16bf cvt_frag(const RawFrag& x) {
  v16bf f;
#pragma unroll
  for (int i = 0; i < 4; ++i)
#pragma unroll
    for (int c = 0; c < 4; ++c) f[4 * i + c] = (__bf16)x.r[i][c];
  return f;
}
__device__ __forceinline__ v16bf load_fragA_g(const float* __restrict__ p, int ld, int lane, int krem) {
  const float* q = p + (long)(lane & 15) * ld;
  int c0 = (lane >> 4) << 3;
  v16bf f;
#pragma unroll
  for (int h = 0; h < 8; ++h) { int c = c0 + h;      f[h]     = (__bf16)((c < krem) ? q[c] : 0.0f); }
#pragma unroll
  for (int h = 0; h < 8; ++h) { int c = c0 + 16 + h; f[h + 8] = (__bf16)((c < krem) ? q[c] : 0.0f); }
  return f;
}
__device__ __forceinline__ v16bf load_fragB_g(const float* __restrict__ p, int ld, int lane, int krem) {
  const float* q = p + (long)(lane & 15) * ld;
  int c0 = (lane >> 4) << 3;
  v16bf f;
#pragma unroll
  for (int h = 0; h < 16; ++h) { int c = c0 + h + ((h >> 3) << 3); f[h] = (__bf16)((c < krem) ? q[c] : 0.0f); }
  return f;
}

template <int TN>
__device__ __forceinline__ void gemm_wave_f32(const float* __restrict__ A, const float* __restrict__ W,
                                              float* __restrict__ C, int K, int lda, int ldw,
                                              int ldc, int ty, int tx0, int lane) {
  v8f acc[TN];
#pragma unroll
  for (int j = 0; j < TN; ++j)
#pragma unroll
    for (int r = 0; r < 8; ++r) acc[j][r] = 0.0f;

  const float* Arow = A + (long)ty * 16 * lda;
  int k = 0;
  for (; k + 32 <= K; k += 32) {
    RawFrag aR = loadA_raw(Arow + k, lda, lane);
    RawFrag bR[TN];
#pragma unroll
    for (int j = 0; j < TN; ++j)
      bR[j] = loadB_raw(W + (long)(tx0 + j) * 16 * ldw + k, ldw, lane);
    v16bf af = cvt_frag(aR);
#pragma unroll
    for (int j = 0; j < TN; ++j) {
      v16bf bf = cvt_frag(bR[j]);
      acc[j] = __builtin_amdgcn_wmma_f32_16x16x32_bf16(false, af, false, bf,
                                                       (short)0, acc[j], false, false);
    }
  }
  if (k < K) {  // K tail (enc layer0 K=300)
    int rem = K - k;
    v16bf af = load_fragA_g(Arow + k, lda, lane, rem);
#pragma unroll
    for (int j = 0; j < TN; ++j) {
      v16bf bf = load_fragB_g(W + (long)(tx0 + j) * 16 * ldw + k, ldw, lane, rem);
      acc[j] = __builtin_amdgcn_wmma_f32_16x16x32_bf16(false, af, false, bf,
                                                       (short)0, acc[j], false, false);
    }
  }
  int n0 = lane & 15;
  int mh = (lane >> 4) << 3;
#pragma unroll
  for (int j = 0; j < TN; ++j) {
    int ncol = (tx0 + j) * 16 + n0;
#pragma unroll
    for (int r = 0; r < 8; ++r) {
      int m = ty * 16 + mh + r;
      C[(long)m * ldc + ncol] = acc[j][r];
    }
  }
}

__global__ void gemm_f32_xwT(const float* __restrict__ A, const float* __restrict__ W,
                             float* __restrict__ C, int M, int N, int K,
                             int lda, int ldw, int ldc) {
  int lane = threadIdx.x & 31;
  int wave = blockIdx.x * (blockDim.x >> 5) + (threadIdx.x >> 5);
  int ntx = N >> 6;
  int tx0 = (wave % ntx) * 4;
  int ty  = wave / ntx;
  if (ty * 16 >= M) return;
  gemm_wave_f32<4>(A, W, C, K, lda, ldw, ldc, ty, tx0, lane);
}

// ================= recurrent / elementwise kernels =================
// Encoder step: gates[d] = h_bf[d] @ Whh_bf[d].T + xproj[d][s_orig] + bsum[d]
// blockIdx.y = direction; 8 blocks x 128 thr per dir (32 waves, N=2048).
__global__ void enc_step(const __bf16* __restrict__ hbf, const __bf16* __restrict__ Whh,
                         const float* __restrict__ xproj, const float* __restrict__ bsum,
                         float* __restrict__ gates, int t) {
  int d = blockIdx.y;
  int lane = threadIdx.x & 31;
  int wave = blockIdx.x * (blockDim.x >> 5) + (threadIdx.x >> 5);
  int tx0 = wave * 4;
  int s_orig = d ? (WS - 1 - t) : t;
  const __bf16* A  = hbf + (long)d * WB * WEH;
  const __bf16* Wd = Whh + (long)d * (4 * WEH) * WEH;
  const float*  Ci = xproj + ((long)d * WS + s_orig) * WB * (4 * WEH);
  gemm_wave_bf<4>(A, Wd, Ci, bsum + d * 4 * WEH, gates + (long)d * WB * 4 * WEH, nullptr,
                  WEH, WEH, WEH, 4 * WEH, 4 * WEH, 0, tx0, lane, 0);
}

// Encoder LSTM elementwise (both dirs); h written f32 + bf16; y[(s_orig, b, d*EH+j)]
__global__ void enc_cell(const float* __restrict__ gates, float* __restrict__ h,
                         __bf16* __restrict__ hbf, float* __restrict__ c,
                         float* __restrict__ y, int t) {
  int idx = blockIdx.x * blockDim.x + threadIdx.x;
  if (idx >= 2 * WB * WEH) return;
  int d = idx / (WB * WEH), r = idx % (WB * WEH);
  int b = r / WEH, j = r % WEH;
  const float* g = gates + ((long)d * WB + b) * 4 * WEH;
  float gi = sig_(g[j]), gf = sig_(g[WEH + j]);
  float gg = tanhf(g[2 * WEH + j]), go = sig_(g[3 * WEH + j]);
  float cv = gf * c[idx] + gi * gg;
  c[idx] = cv;
  float hv = go * tanhf(cv);
  h[idx] = hv;
  hbf[idx] = (__bf16)hv;
  int s_orig = d ? (WS - 1 - t) : t;
  y[((long)s_orig * WB + b) * WH + d * WEH + j] = hv;
}

// Decoder LSTM elementwise (H=1024); h written f32 + bf16
__global__ void dec_cell(const float* __restrict__ gates, float* __restrict__ h,
                         __bf16* __restrict__ hbf, float* __restrict__ c) {
  int idx = blockIdx.x * blockDim.x + threadIdx.x;
  if (idx >= WB * WH) return;
  int b = idx >> 10, j = idx & (WH - 1);
  const float* g = gates + (long)b * 4 * WH;
  float gi = sig_(g[j]), gf = sig_(g[WH + j]);
  float gg = tanhf(g[2 * WH + j]), go = sig_(g[3 * WH + j]);
  float cv = gf * c[idx] + gi * gg;
  c[idx] = cv;
  float hv = go * tanhf(cv);
  h[idx] = hv;
  hbf[idx] = (__bf16)hv;
}

// scores[b][s] = dot(y1[s,b,:], q[b,:]) ; mask -> NEG. One wave per (b,s).
__global__ void attn_scores(const float* __restrict__ y1, const float* __restrict__ q,
                            const unsigned char* __restrict__ mask, float* __restrict__ w) {
  int wave = blockIdx.x * (blockDim.x >> 5) + (threadIdx.x >> 5);
  int lane = threadIdx.x & 31;
  if (wave >= WB * WS) return;
  int b = wave >> 8, s = wave & (WS - 1);
  const float* e  = y1 + ((long)s * WB + b) * WH;
  const float* qq = q + (long)b * WH;
  float sum = 0.0f;
  for (int h = lane; h < WH; h += 32) sum += e[h] * qq[h];
#pragma unroll
  for (int off = 16; off > 0; off >>= 1) sum += __shfl_xor(sum, off, 32);
  if (lane == 0) w[b * WS + s] = mask[b * WS + s] ? NEGV : sum;
}

// softmax over S=256 per row, one block (256 threads) per b
__global__ void attn_softmax(float* __restrict__ w) {
  __shared__ float redm[8];
  __shared__ float reds[8];
  int b = blockIdx.x, tid = threadIdx.x;
  float v = w[b * WS + tid];
  float m = v;
#pragma unroll
  for (int off = 16; off > 0; off >>= 1) m = fmaxf(m, __shfl_xor(m, off, 32));
  if ((tid & 31) == 0) redm[tid >> 5] = m;
  __syncthreads();
  m = redm[0];
#pragma unroll
  for (int i = 1; i < 8; ++i) m = fmaxf(m, redm[i]);
  float e = expf(v - m);
  float s = e;
#pragma unroll
  for (int off = 16; off > 0; off >>= 1) s += __shfl_xor(s, off, 32);
  if ((tid & 31) == 0) reds[tid >> 5] = s;
  __syncthreads();
  s = 0.0f;
#pragma unroll
  for (int i = 0; i < 8; ++i) s += reds[i];
  w[b * WS + tid] = e / s;
}

// ctx[b][h] = sum_s a[b][s] * y1[s,b,h]
__global__ void attn_ctx(const float* __restrict__ a, const float* __restrict__ y1,
                         float* __restrict__ ctx) {
  int idx = blockIdx.x * blockDim.x + threadIdx.x;
  if (idx >= WB * WH) return;
  int b = idx >> 10, h = idx & (WH - 1);
  const float* ab = a + b * WS;
  const float* e  = y1 + (long)b * WH + h;
  float s = 0.0f;
  for (int si = 0; si < WS; ++si) s += ab[si] * e[(long)si * WB * WH];
  ctx[idx] = s;
}

// x[b] = [emb(tok), op[b]]  (16 x 2048) -> bf16 (GEMM A operand)
__global__ void dec_pack_x(const int* __restrict__ toks, const float* __restrict__ emb,
                           const float* __restrict__ op, __bf16* __restrict__ x, int t) {
  int idx = blockIdx.x * blockDim.x + threadIdx.x;
  if (idx >= WB * 2 * WH) return;
  int b = idx / (2 * WH), j = idx % (2 * WH);
  float v;
  if (j < WH) {
    int tok = (t == 0) ? 3 : toks[b * WT + (t - 1)];
    v = emb[(long)tok * WH + j];
  } else {
    v = op[(long)b * WH + (j - WH)];
  }
  x[idx] = (__bf16)v;
}

// x[b] = [a[b] (n1), b2[b] (n2)] -> bf16
__global__ void pack2(const float* __restrict__ a, const float* __restrict__ b2,
                      __bf16* __restrict__ x, int n1, int n2) {
  int n = n1 + n2;
  int idx = blockIdx.x * blockDim.x + threadIdx.x;
  if (idx >= WB * n) return;
  int b = idx / n, j = idx % n;
  x[idx] = (__bf16)((j < n1) ? a[(long)b * n1 + j] : b2[(long)b * n2 + (j - n1)]);
}

// hdec[l][b][d*EH+jj] = fin[l][d][b][jj]
__global__ void pack_dec_init(const float* __restrict__ fin, float* __restrict__ dst) {
  int idx = blockIdx.x * blockDim.x + threadIdx.x;
  if (idx >= 2 * WB * WH) return;
  int l = idx / (WB * WH), r = idx % (WB * WH);
  int b = r / WH, j = r % WH;
  int d = j / WEH, jj = j % WEH;
  dst[idx] = fin[(((long)(l * 2 + d) * WB) + b) * WEH + jj];
}

// inp (B,S,300) -> xs (S*B, 300) with row m = s*B+b
__global__ void transpose_inp(const float* __restrict__ inp, float* __restrict__ xs) {
  int idx = blockIdx.x * blockDim.x + threadIdx.x;
  if (idx >= WS * WB * 300) return;
  int m = idx / 300, j = idx % 300;
  int s = m >> 4, b = m & 15;
  xs[idx] = inp[((long)b * WS + s) * 300 + j];
}

// f32 -> bf16 bulk conversion (4 elems/thread, vectorized)
__global__ void cvt_bf16(const float* __restrict__ a, __bf16* __restrict__ b, long n) {
  long i = ((long)blockIdx.x * blockDim.x + threadIdx.x) * 4;
  if (i >= n) return;
  v4f v = *(const v4f*)(a + i);
  v4bf o;
#pragma unroll
  for (int c = 0; c < 4; ++c) o[c] = (__bf16)v[c];
  *(v4bf*)(b + i) = o;
}

__global__ void vec_add(const float* a, const float* b, float* c, int n) {
  int i = blockIdx.x * blockDim.x + threadIdx.x;
  if (i < n) c[i] = a[i] + b[i];
}
__global__ void vec_copy(const float* a, float* b, int n) {
  int i = blockIdx.x * blockDim.x + threadIdx.x;
  if (i < n) b[i] = a[i];
}
__global__ void vec_zero(float* a, int n) {
  int i = blockIdx.x * blockDim.x + threadIdx.x;
  if (i < n) a[i] = 0.0f;
}
__global__ void vec_zero_bf(__bf16* a, int n) {
  int i = blockIdx.x * blockDim.x + threadIdx.x;
  if (i < n) a[i] = (__bf16)0.0f;
}

// ------------------------------------------------------------------
extern "C" void kernel_launch(void* const* d_in, const int* in_sizes, int n_in,
                              void* d_out, int out_size, void* d_ws, size_t ws_size,
                              hipStream_t stream) {
  const float* inp   = (const float*)d_in[0];
  const unsigned char* mask = (const unsigned char*)d_in[1];
  const int*   toks  = (const int*)d_in[2];
  const float* eWih0 = (const float*)d_in[3];
  const float* eWhh0 = (const float*)d_in[4];
  const float* eBih0 = (const float*)d_in[5];
  const float* eBhh0 = (const float*)d_in[6];
  const float* eWih1 = (const float*)d_in[7];
  const float* eWhh1 = (const float*)d_in[8];
  const float* eBih1 = (const float*)d_in[9];
  const float* eBhh1 = (const float*)d_in[10];
  const float* dWih0 = (const float*)d_in[11];
  const float* dWhh0 = (const float*)d_in[12];
  const float* dBih0 = (const float*)d_in[13];
  const float* dBhh0 = (const float*)d_in[14];
  const float* dWih1 = (const float*)d_in[15];
  const float* dWhh1 = (const float*)d_in[16];
  const float* dBih1 = (const float*)d_in[17];
  const float* dBhh1 = (const float*)d_in[18];
  const float* demb  = (const float*)d_in[19];
  const float* linW  = (const float*)d_in[20];
  const float* linb  = (const float*)d_in[21];
  const float* genW  = (const float*)d_in[22];
  const float* genb  = (const float*)d_in[23];
  float* out = (float*)d_out;

  // -------- workspace carve-up: f32 region then bf16 region --------
  float* ws = (float*)d_ws;
  size_t o = 0;
  auto carve = [&](size_t n) { float* p = ws + o; o += n; return p; };
  float* xs     = carve((size_t)WS * WB * 300);        // 1.23M
  float* xproj  = carve((size_t)2 * WS * WB * 2048);   // 16.8M
  float* y0     = carve((size_t)WS * WB * WH);         // 4.2M
  float* y1     = carve((size_t)WS * WB * WH);         // 4.2M
  float* hst    = carve(2 * WB * WEH);
  float* cst    = carve(2 * WB * WEH);
  float* hfin   = carve(2 * 2 * WB * WEH);
  float* cfin   = carve(2 * 2 * WB * WEH);
  float* hdec   = carve(2 * WB * WH);
  float* cdec   = carve(2 * WB * WH);
  float* genc   = carve(2 * WB * 2048);
  float* gdec   = carve(WB * 4096);
  float* opbuf  = carve(WB * WH);
  float* ctx    = carve(WB * WH);
  float* scores = carve(WB * WS);
  float* bsE0   = carve(2 * 2048);
  float* bsE1   = carve(2 * 2048);
  float* bsD0   = carve(4096);
  float* bsD1   = carve(4096);
  o = (o + 7) & ~(size_t)7;  // 32B-align the bf16 region
  __bf16* bfbase = (__bf16*)(ws + o);
  size_t ob = 0;
  auto carveb = [&](size_t n) { __bf16* p = bfbase + ob; ob += (n + 15) & ~(size_t)15; return p; };
  __bf16* eWhh0_b = carveb((size_t)2 * 4 * WEH * WEH);
  __bf16* eWhh1_b = carveb((size_t)2 * 4 * WEH * WEH);
  __bf16* dWih0_b = carveb((size_t)4 * WH * 2 * WH);
  __bf16* dWhh0_b = carveb((size_t)4 * WH * WH);
  __bf16* dWih1_b = carveb((size_t)4 * WH * WH);
  __bf16* dWhh1_b = carveb((size_t)4 * WH * WH);
  __bf16* linW_b  = carveb((size_t)WH * 2 * WH);
  __bf16* genW_b  = carveb((size_t)WV * WH);
  __bf16* hst_b   = carveb(2 * WB * WEH);
  __bf16* hdec_b  = carveb(2 * WB * WH);
  __bf16* xdec_b  = carveb(WB * 2 * WH);
  __bf16* x2_b    = carveb(WB * 2 * WH);
  __bf16* op_b    = carveb(WB * WH);

  auto gemmb = [&](const __bf16* A, const __bf16* W, const float* Cin, const float* bias,
                   float* C, __bf16* Cbf, int M, int N, int K, int lda, int ldw,
                   int ldcin, int ldc, int ep) {
    int waves = (M / 16) * (N / 64);
    gemm_bf16_xwT<<<(waves + 3) / 4, 128, 0, stream>>>(A, W, Cin, bias, C, Cbf,
                                                       M, N, K, lda, ldw, ldcin, ldc, ep);
  };
  auto gemmf = [&](const float* A, const float* W, float* C, int M, int N, int K,
                   int lda, int ldw, int ldc) {
    int waves = (M / 16) * (N / 64);
    gemm_f32_xwT<<<(waves + 3) / 4, 128, 0, stream>>>(A, W, C, M, N, K, lda, ldw, ldc);
  };
  auto cv = [&](const float* a, __bf16* b, long n) {
    cvt_bf16<<<(unsigned)((n / 4 + 255) / 256), 256, 0, stream>>>(a, b, n);
  };
  auto E = [&](int n) { return dim3((n + 255) / 256); };

  // -------- one-time per call: weight bf16 staging + combined biases --------
  cv(eWhh0, eWhh0_b, (long)2 * 4 * WEH * WEH);
  cv(eWhh1, eWhh1_b, (long)2 * 4 * WEH * WEH);
  cv(dWih0, dWih0_b, (long)4 * WH * 2 * WH);
  cv(dWhh0, dWhh0_b, (long)4 * WH * WH);
  cv(dWih1, dWih1_b, (long)4 * WH * WH);
  cv(dWhh1, dWhh1_b, (long)4 * WH * WH);
  cv(linW,  linW_b,  (long)WH * 2 * WH);
  cv(genW,  genW_b,  (long)WV * WH);
  vec_add<<<E(2 * 2048), 256, 0, stream>>>(eBih0, eBhh0, bsE0, 2 * 2048);
  vec_add<<<E(2 * 2048), 256, 0, stream>>>(eBih1, eBhh1, bsE1, 2 * 2048);
  vec_add<<<E(4096), 256, 0, stream>>>(dBih0, dBhh0, bsD0, 4096);
  vec_add<<<E(4096), 256, 0, stream>>>(dBih1, dBhh1, bsD1, 4096);

  // -------- encoder layer 0 --------
  transpose_inp<<<E(WS * WB * 300), 256, 0, stream>>>(inp, xs);
  for (int d = 0; d < 2; ++d)  // xproj[d] = xs @ Wih0[d].T   (4096 x 2048, K=300)
    gemmf(xs, eWih0 + (size_t)d * 2048 * 300,
          xproj + (size_t)d * WS * WB * 2048, WS * WB, 2048, 300, 300, 300, 2048);
  vec_zero<<<E(2 * WB * WEH), 256, 0, stream>>>(hst, 2 * WB * WEH);
  vec_zero<<<E(2 * WB * WEH), 256, 0, stream>>>(cst, 2 * WB * WEH);
  vec_zero_bf<<<E(2 * WB * WEH), 256, 0, stream>>>(hst_b, 2 * WB * WEH);
  for (int t = 0; t < WS; ++t) {
    enc_step<<<dim3(8, 2), 128, 0, stream>>>(hst_b, eWhh0_b, xproj, bsE0, genc, t);
    enc_cell<<<E(2 * WB * WEH), 256, 0, stream>>>(genc, hst, hst_b, cst, y0, t);
  }
  vec_copy<<<E(2 * WB * WEH), 256, 0, stream>>>(hst, hfin, 2 * WB * WEH);
  vec_copy<<<E(2 * WB * WEH), 256, 0, stream>>>(cst, cfin, 2 * WB * WEH);

  // -------- encoder layer 1 --------
  for (int d = 0; d < 2; ++d)  // xproj[d] = y0 @ Wih1[d].T   (4096 x 2048, K=1024)
    gemmf(y0, eWih1 + (size_t)d * 2048 * WH,
          xproj + (size_t)d * WS * WB * 2048, WS * WB, 2048, WH, WH, WH, 2048);
  vec_zero<<<E(2 * WB * WEH), 256, 0, stream>>>(hst, 2 * WB * WEH);
  vec_zero<<<E(2 * WB * WEH), 256, 0, stream>>>(cst, 2 * WB * WEH);
  vec_zero_bf<<<E(2 * WB * WEH), 256, 0, stream>>>(hst_b, 2 * WB * WEH);
  for (int t = 0; t < WS; ++t) {
    enc_step<<<dim3(8, 2), 128, 0, stream>>>(hst_b, eWhh1_b, xproj, bsE1, genc, t);
    enc_cell<<<E(2 * WB * WEH), 256, 0, stream>>>(genc, hst, hst_b, cst, y1, t);
  }
  vec_copy<<<E(2 * WB * WEH), 256, 0, stream>>>(hst, hfin + 2 * WB * WEH, 2 * WB * WEH);
  vec_copy<<<E(2 * WB * WEH), 256, 0, stream>>>(cst, cfin + 2 * WB * WEH, 2 * WB * WEH);

  // -------- decoder init --------
  pack_dec_init<<<E(2 * WB * WH), 256, 0, stream>>>(hfin, hdec);
  pack_dec_init<<<E(2 * WB * WH), 256, 0, stream>>>(cfin, cdec);
  cv(hdec, hdec_b, 2 * WB * WH);
  vec_zero<<<E(WB * WH), 256, 0, stream>>>(opbuf, WB * WH);

  float*  h0  = hdec;            float* c0 = cdec;
  float*  h1  = hdec + WB * WH;  float* c1 = cdec + WB * WH;
  __bf16* h0b = hdec_b;
  __bf16* h1b = hdec_b + WB * WH;

  // -------- decoder steps --------
  for (int t = 0; t < WT; ++t) {
    dec_pack_x<<<E(WB * 2 * WH), 256, 0, stream>>>(toks, demb, opbuf, xdec_b, t);
    // layer 0: gates = x @ Wih0.T + bsum ; += h0 @ Whh0.T
    gemmb(xdec_b, dWih0_b, nullptr, bsD0, gdec, nullptr,
          WB, 4 * WH, 2 * WH, 2 * WH, 2 * WH, 0, 4 * WH, 0);
    gemmb(h0b, dWhh0_b, gdec, nullptr, gdec, nullptr,
          WB, 4 * WH, WH, WH, WH, 4 * WH, 4 * WH, 0);
    dec_cell<<<E(WB * WH), 256, 0, stream>>>(gdec, h0, h0b, c0);
    // layer 1
    gemmb(h0b, dWih1_b, nullptr, bsD1, gdec, nullptr,
          WB, 4 * WH, WH, WH, WH, 0, 4 * WH, 0);
    gemmb(h1b, dWhh1_b, gdec, nullptr, gdec, nullptr,
          WB, 4 * WH, WH, WH, WH, 4 * WH, 4 * WH, 0);
    dec_cell<<<E(WB * WH), 256, 0, stream>>>(gdec, h1, h1b, c1);
    // attention
    attn_scores<<<(WB * WS) / 8, 256, 0, stream>>>(y1, h1, mask, scores);
    attn_softmax<<<WB, WS, 0, stream>>>(scores);
    attn_ctx<<<E(WB * WH), 256, 0, stream>>>(scores, y1, ctx);
    // opn = tanh([ctx, decout] @ linW.T + linb) ; f32 for next-step op, bf16 for gen GEMM
    pack2<<<E(WB * 2 * WH), 256, 0, stream>>>(ctx, h1, x2_b, WH, WH);
    gemmb(x2_b, linW_b, nullptr, linb, opbuf, op_b,
          WB, WH, 2 * WH, 2 * WH, 2 * WH, 0, WH, 1);
    // logits -> d_out[b][t][:]
    gemmb(op_b, genW_b, nullptr, genb, out + (size_t)t * WV, nullptr,
          WB, WV, WH, WH, WH, 0, WT * WV, 0);
  }
}
